// MultiViewNetwork_66176856097042
// MI455X (gfx1250) — matrix operand order repaired and checked by reading, not compiled, not run on hardware
//
#include <hip/hip_runtime.h>
#include <hip/hip_bf16.h>

// ---------------------------------------------------------------------------
// Types for CDNA5 WMMA (wave32, 16x16x32 bf16 -> f32)
// ---------------------------------------------------------------------------
typedef __attribute__((ext_vector_type(16))) __bf16 bf16x16;
typedef __attribute__((ext_vector_type(8)))  float  f32x8;

__device__ __forceinline__ float bf2f(unsigned short h) {
  return __uint_as_float(((unsigned)h) << 16);
}
__device__ __forceinline__ unsigned short f2bf(float f) {
  unsigned u = __float_as_uint(f);
  u += 0x7FFFu + ((u >> 16) & 1u);   // round-to-nearest-even
  return (unsigned short)(u >> 16);
}

// ---------------------------------------------------------------------------
// Elementwise conversion kernels
// ---------------------------------------------------------------------------
__global__ void k_f32_to_bf16(const float* __restrict__ x, unsigned short* __restrict__ y, long n) {
  long i = (long)blockIdx.x * 256 + threadIdx.x;
  if (i < n) y[i] = f2bf(x[i]);
}
__global__ void k_bf16_to_f32(const unsigned short* __restrict__ x, float* __restrict__ y, long n) {
  long i = (long)blockIdx.x * 256 + threadIdx.x;
  if (i < n) y[i] = bf2f(x[i]);
}
// OIHW f32 -> [Cout][K] bf16 with K-order k = (kh*ks+kw)*Cin + ci
// (tap-outer / channel-inner: a 16-aligned K-run never crosses a tap, and
//  rows are contiguous so the B tile can be DMA'd with async b64 loads)
__global__ void k_quant_w(const float* __restrict__ w, unsigned short* __restrict__ q,
                          int cout, int cin, int ks) {
  long n = (long)cout * cin * ks * ks;
  long i = (long)blockIdx.x * 256 + threadIdx.x;
  if (i >= n) return;
  int KK = ks * ks;
  int CK = cin * KK;                       // == K
  int o   = (int)(i / CK);
  int rem = (int)(i - (long)o * CK);       // == ci*KK + (kh*ks+kw)
  int ci  = rem / KK;
  int r   = rem - ci * KK;                 // tap index kh*ks+kw
  q[(size_t)o * CK + (size_t)r * cin + ci] = f2bf(w[i]);
}
// y = bf16( bf16(a) + f32 b )   (pre-fuse residual add with deep_128)
__global__ void k_add_f32res(const unsigned short* __restrict__ a, const float* __restrict__ b,
                             unsigned short* __restrict__ y, long n) {
  long i = (long)blockIdx.x * 256 + threadIdx.x;
  if (i < n) y[i] = f2bf(bf2f(a[i]) + b[i]);
}

// ---------------------------------------------------------------------------
// Implicit-GEMM conv with WMMA bf16 -> f32, fused scale/bias/ReLU/residual.
//   X : [Cin][Hin][Win] bf16      Wq: [Cout][K] bf16, K=(kh*ks+kw)*Cin+ci
//   Y : [Cout][Hout][Wout] bf16   resid (optional): same layout as Y
// Block = 256 threads = 8 waves; block tile M=128 pixels x N=32 channels;
// each wave owns one 16-row A fragment and computes two 16x16 WMMAs.
// Double-buffered LDS; B tile staged with global_load_async_to_lds_b64 and
// pipelined one K-step ahead; single s_wait_asynccnt + barrier per step.
// Requires: Hout*Wout % 128 == 0, Cout % 32 == 0, Cin % 32 == 0.
// ---------------------------------------------------------------------------
#define LDA 36             // halves; 72 B rows -> 8B-aligned b64 stores
#define LDB 36
#define ATILE (128 * LDA)
#define BTILE (32 * LDB)

template <int KS>
__global__ void __launch_bounds__(256)
k_conv_igemm_wmma(const unsigned short* __restrict__ X, int Cin, int Hin, int Win,
                  const unsigned short* __restrict__ Wq,
                  const float* __restrict__ scale, const float* __restrict__ bias,
                  unsigned short* __restrict__ Y, int Cout, int Hout, int Wout,
                  int stride, int pad,
                  const unsigned short* __restrict__ resid, int relu) {
  constexpr int KK = KS * KS;
  __shared__ unsigned short As[2 * ATILE];   // [buf][m][k] im2col tiles
  __shared__ unsigned short Bs[2 * BTILE];   // [buf][n][k] weight tiles

  const int m0   = blockIdx.x * 128;
  const int n0   = blockIdx.y * 32;
  const int t    = threadIdx.x;
  const int lane = t & 31;
  const int wave = t >> 5;                   // 8 waves = 8 M-subtiles
  const int K    = Cin * KK;
  const unsigned HW = (unsigned)Hin * (unsigned)Win;

  // --- A staging: each thread stages 1 row(m) x 16 consecutive k's ---
  const int a_m = t >> 1;                    // 0..127
  const int a_k = (t & 1) * 16;              // 0 or 16
  const int gm  = m0 + a_m;
  const int oh  = gm / Wout;
  const int ow  = gm - oh * Wout;
  const int ohs = oh * stride - pad;
  const int ows = ow * stride - pad;
  // rolling state for k = k0 + a_k in tap-outer order: k = (kh*KS+kw)*Cin + ci0
  // (a_k < 32 <= Cin, so the initial tap is (0,0))
  int kh = 0, kw = 0, ci0 = a_k;

  // --- B staging: each thread async-copies one b64 (4 halves) per step ---
  const int b_n = t >> 3;                    // 0..31
  const int b_k = (t & 7) * 4;               // 0,4,...,28

  f32x8 acc0 = {0.f, 0.f, 0.f, 0.f, 0.f, 0.f, 0.f, 0.f};
  f32x8 acc1 = {0.f, 0.f, 0.f, 0.f, 0.f, 0.f, 0.f, 0.f};

  const int l16  = lane & 15;
  const int half = lane >> 4;

  auto stage = [&](int buf, int k0) {
    // ---- B tile: async DMA global -> LDS (2 KB tile, ASYNCcnt) ----
    const unsigned short* gsrc = Wq + (size_t)(n0 + b_n) * K + (k0 + b_k);
    unsigned blds = (unsigned)(unsigned long long)(&Bs[buf * BTILE + b_n * LDB + b_k]);
    asm volatile("global_load_async_to_lds_b64 %0, %1, off"
                 :: "v"(blds), "v"((unsigned long long)gsrc) : "memory");
    if (k0 + 32 < K)   // pull the following chunk toward L2
      __builtin_prefetch(gsrc + 32, 0, 1);

    // ---- A tile: tap is constant for the whole 16-element run ----
    const int ih = ohs + kh;
    const int iw = ows + kw;
    const bool ok =
        ((unsigned)ih < (unsigned)Hin) & ((unsigned)iw < (unsigned)Win);
    unsigned idx = (unsigned)ci0 * HW + (unsigned)ih * (unsigned)Win + (unsigned)iw;
    idx = ok ? idx : 0u;                      // safe base when out of bounds
    const unsigned msk = ok ? 0xFFFFu : 0u;
    union { unsigned short h[16]; unsigned long long q[4]; } av;
#pragma unroll
    for (int i = 0; i < 16; ++i) {            // 1 load + 1 add + 1 and per elem
      av.h[i] = (unsigned short)((unsigned)X[idx] & msk);
      idx += HW;
    }
    unsigned long long* dst =
        (unsigned long long*)&As[buf * ATILE + a_m * LDA + a_k];
#pragma unroll
    for (int qq = 0; qq < 4; ++qq) dst[qq] = av.q[qq];   // -> ds_store_2addr_b64

    // advance k by 32: at most one tap carry since 32 <= Cin
    ci0 += 32;
    if (KS > 1) {
      const bool wrap = ci0 >= Cin;
      ci0 = wrap ? ci0 - Cin : ci0;
      int kwn = wrap ? kw + 1 : kw;
      const bool wrap2 = kwn == KS;
      kw = wrap2 ? 0 : kwn;
      kh = wrap ? (wrap2 ? kh + 1 : kh) : kh;
    }
  };

  // ---- pipelined main loop: stage s+1 while computing s ----
  stage(0, 0);
  asm volatile("s_wait_asynccnt 0x0" ::: "memory");
  __syncthreads();

  const int nsteps = K >> 5;
  for (int s = 0; s < nsteps; ++s) {
    const int p = s & 1;
    if (s + 1 < nsteps) stage(1 - p, (s + 1) << 5);

    // ---- fragment loads in contiguous k order (k = 16*half + s), same for A and B ----
    union { bf16x16 v; unsigned u[8]; } Af, Bf0, Bf1;
    const unsigned short* ap = &As[p * ATILE + (wave * 16 + l16) * LDA];
#pragma unroll
    for (int j = 0; j < 8; ++j) {
      // A, contiguous k order: lanes0-15 K=0-15; lanes16-31 K=16-31 (matches B)
      int kp = 2 * j + half * 16;
      Af.u[j] = *(const unsigned*)(ap + kp);
    }
    const unsigned short* bp0 = &Bs[p * BTILE + l16 * LDB + half * 16];
    const unsigned short* bp1 = bp0 + 16 * LDB;
#pragma unroll
    for (int j = 0; j < 8; ++j) {
      Bf0.u[j] = *(const unsigned*)(bp0 + 2 * j);
      Bf1.u[j] = *(const unsigned*)(bp1 + 2 * j);
    }
    acc0 = __builtin_amdgcn_wmma_f32_16x16x32_bf16(false, Af.v, false, Bf0.v,
                                                   (short)0, acc0, false, false);
    acc1 = __builtin_amdgcn_wmma_f32_16x16x32_bf16(false, Af.v, false, Bf1.v,
                                                   (short)0, acc1, false, false);
    asm volatile("s_wait_asynccnt 0x0" ::: "memory");
    __syncthreads();
  }

  // ---- epilogue: BN scale/bias, residual, ReLU, store bf16 ----
  const size_t M = (size_t)Hout * Wout;
#pragma unroll
  for (int nh = 0; nh < 2; ++nh) {
    const int col = n0 + nh * 16 + l16;
    const float s = scale[col];
    const float b = bias[col];
    const f32x8& acc = nh ? acc1 : acc0;
#pragma unroll
    for (int r = 0; r < 8; ++r) {
      int row = m0 + wave * 16 + r + half * 8;   // C/D: VGPR r -> M=r (+8 hi lanes)
      float v = acc[r] * s + b;
      if (resid) v += bf2f(resid[(size_t)col * M + row]);
      if (relu && v < 0.f) v = 0.f;
      Y[(size_t)col * M + row] = f2bf(v);
    }
  }
}

// ---------------------------------------------------------------------------
// Bilinear gather (matches reference's clamped sampling)
// ---------------------------------------------------------------------------
__device__ __forceinline__ float bilin(const unsigned short* __restrict__ fc,
                                       int H, int W, float h, float w) {
  float h0f = floorf(h), w0f = floorf(w);
  float fh = h - h0f, fw = w - w0f;
  int h0 = (int)h0f; h0 = h0 < 0 ? 0 : (h0 > H - 1 ? H - 1 : h0);
  int h1 = h0 + 1;   h1 = h1 > H - 1 ? H - 1 : h1;
  int w0 = (int)w0f; w0 = w0 < 0 ? 0 : (w0 > W - 1 ? W - 1 : w0);
  int w1 = w0 + 1;   w1 = w1 > W - 1 ? W - 1 : w1;
  float g00 = bf2f(fc[(size_t)h0 * W + w0]);
  float g01 = bf2f(fc[(size_t)h0 * W + w1]);
  float g10 = bf2f(fc[(size_t)h1 * W + w0]);
  float g11 = bf2f(fc[(size_t)h1 * W + w1]);
  return g00 * (1.f - fh) * (1.f - fw) + g01 * (1.f - fh) * fw +
         g10 * fh * (1.f - fw) + g11 * fh * fw;
}

// Fused grid_to_point + voxel scatter-max; grid = (ceil(N/256), C)
#define ENC_NEGINF 0x007FFFFFu   // encode(-inf)
__global__ void k_gather_scatter_max(const unsigned short* __restrict__ feat, int H, int W,
                                     const float* __restrict__ cA, float srA,
                                     const float* __restrict__ cB, float srB,
                                     int Ho, int Wo, unsigned* __restrict__ enc, int N) {
  int n = blockIdx.x * 256 + threadIdx.x;
  if (n >= N) return;
  int c = blockIdx.y;
  int hi = (int)floorf(cB[2 * n] * srB);
  int wi = (int)floorf(cB[2 * n + 1] * srB);
  if (hi < 0 || hi >= Ho || wi < 0 || wi >= Wo) return;
  float v = bilin(feat + (size_t)c * H * W, H, W, cA[2 * n] * srA, cA[2 * n + 1] * srA);
  unsigned u = __float_as_uint(v);
  unsigned key = (u & 0x80000000u) ? ~u : (u | 0x80000000u);
  atomicMax(&enc[(size_t)c * Ho * Wo + (size_t)hi * Wo + wi], key);
}
__global__ void k_pool_init(unsigned* __restrict__ enc, long n) {
  long i = (long)blockIdx.x * 256 + threadIdx.x;
  if (i < n) enc[i] = ENC_NEGINF;
}
__global__ void k_pool_decode(const unsigned* __restrict__ enc, unsigned short* __restrict__ out, long n) {
  long i = (long)blockIdx.x * 256 + threadIdx.x;
  if (i >= n) return;
  unsigned key = enc[i];
  float f = 0.f;
  if (key != ENC_NEGINF) {
    unsigned u = (key & 0x80000000u) ? (key & 0x7FFFFFFFu) : ~key;
    f = __uint_as_float(u);
    if (!isfinite(f)) f = 0.f;   // empty/inf voxels -> 0
  }
  out[i] = f2bf(f);
}

// grid_to_point to f32 output [C][N]; grid = (ceil(N/256), C)
__global__ void k_grid_to_point(const unsigned short* __restrict__ feat, int H, int W,
                                const float* __restrict__ coords, float sr,
                                float* __restrict__ out, int N) {
  int n = blockIdx.x * 256 + threadIdx.x;
  if (n >= N) return;
  int c = blockIdx.y;
  out[(size_t)c * N + n] =
      bilin(feat + (size_t)c * H * W, H, W, coords[2 * n] * sr, coords[2 * n + 1] * sr);
}

// bilinear align_corners=True upsample; grid = (ceil(Ho*Wo/256), C)
__global__ void k_upsample_ac(const unsigned short* __restrict__ in, int H, int W,
                              unsigned short* __restrict__ out, int Ho, int Wo) {
  int r = blockIdx.x * 256 + threadIdx.x;
  if (r >= Ho * Wo) return;
  int c  = blockIdx.y;
  int oh = r / Wo, ow = r - oh * Wo;
  float hs = (Ho > 1) ? oh * (float)(H - 1) / (float)(Ho - 1) : 0.f;
  float ws = (Wo > 1) ? ow * (float)(W - 1) / (float)(Wo - 1) : 0.f;
  out[(size_t)c * Ho * Wo + r] = f2bf(bilin(in + (size_t)c * H * W, H, W, hs, ws));
}

// 1x1 aux head (Cout=3, with bias), bf16 in -> f32 out [3][npix];
// grid = (ceil(npix/256), 3)
__global__ void k_aux(const unsigned short* __restrict__ x, int Cin,
                      const float* __restrict__ w, const float* __restrict__ b,
                      float* __restrict__ out, int npix) {
  int p = blockIdx.x * 256 + threadIdx.x;
  if (p >= npix) return;
  int o = blockIdx.y;
  float acc = b[o];
  for (int c = 0; c < Cin; ++c) acc += w[o * Cin + c] * bf2f(x[(size_t)c * npix + p]);
  out[(size_t)o * npix + p] = acc;
}

// ---------------------------------------------------------------------------
// Host orchestration
// ---------------------------------------------------------------------------
extern "C" void kernel_launch(void* const* d_in, const int* in_sizes, int n_in,
                              void* d_out, int out_size, void* d_ws, size_t ws_size,
                              hipStream_t stream) {
  (void)n_in; (void)out_size; (void)ws_size;
  auto F = [&](int i) -> const float* { return (const float*)d_in[i]; };
  const float* feat_in = F(0);   // [1,192,512,512]
  const float* dcoord  = F(1);   // [1,N,2,1]
  const float* ccoord  = F(2);   // [1,N,2,1]
  const float* deep    = F(3);   // [1,128,64,64]
  const int N = in_sizes[1] / 2;

  // ---- workspace bump allocator ----
  char* base = (char*)d_ws;
  size_t off = 0;
  auto alloc = [&](size_t bytes) -> void* {
    off = (off + 255) & ~(size_t)255;
    void* p = base + off;
    off += bytes;
    return p;
  };
  unsigned short* X0      = (unsigned short*)alloc(192ull * 512 * 512 * 2);
  unsigned short* cat224  = (unsigned short*)alloc(224ull * 65536 * 2); // res0|res1u|res2u
  unsigned short* des0cat = (unsigned short*)alloc(64ull  * 65536 * 2); // des0|c2d0
  unsigned short* tmpA    = (unsigned short*)alloc(128ull * 65536 * 2);
  unsigned short* tmpB    = (unsigned short*)alloc(128ull * 65536 * 2);
  unsigned short* d2c0    = (unsigned short*)alloc(32ull  * 32768 * 2);
  unsigned short* cyl0    = (unsigned short*)alloc(32ull  * 32768 * 2);
  unsigned short* cylT    = (unsigned short*)alloc(64ull  * 32768 * 2);
  unsigned short* des1cat = (unsigned short*)alloc(128ull * 16384 * 2); // des1|c2d1
  unsigned short* des1d   = (unsigned short*)alloc(64ull  * 16384 * 2);
  unsigned short* d2c1    = (unsigned short*)alloc(64ull  * 8192  * 2);
  unsigned short* cyl1    = (unsigned short*)alloc(64ull  * 8192  * 2);
  unsigned short* sum2    = (unsigned short*)alloc(128ull * 4096  * 2);
  unsigned short* des2    = (unsigned short*)alloc(128ull * 4096  * 2);
  unsigned short* desout  = (unsigned short*)alloc(64ull  * 65536 * 2);
  unsigned*       enc     = (unsigned*)alloc(64ull * 65536 * 4);

  auto gb = [](long n) { return (unsigned)((n + 255) / 256); };

  // ---- weight quantization (f32 OIHW -> bf16 [Cout][K], tap-outer K) ----
  auto quantW = [&](const float* w, int cout, int cin, int ks) -> unsigned short* {
    long n = (long)cout * cin * ks * ks;
    unsigned short* q = (unsigned short*)alloc((size_t)n * 2);
    k_quant_w<<<gb(n), 256, 0, stream>>>(w, q, cout, cin, ks);
    return q;
  };
  struct Q3 { unsigned short* w; const float* s; const float* b; };
  // leaf order inside a cbr dict (sorted): b, s, w
  auto mkcbr = [&](int bi, int si, int wi, int cout, int cin, int ks) -> Q3 {
    return { quantW(F(wi), cout, cin, ks), F(si), F(bi) };
  };
  struct QB { Q3 c1, c2; };
  // block dict sorted leaves: b1,b2,s1,s2,w1,w2 at base..base+5
  auto mkblk = [&](int bse, int C) -> QB {
    return { { quantW(F(bse + 4), C, C, 3), F(bse + 2), F(bse + 0) },
             { quantW(F(bse + 5), C, C, 3), F(bse + 3), F(bse + 1) } };
  };
  auto conv = [&](const unsigned short* X, int Cin, int Hin, int Win, const Q3& q,
                  unsigned short* Y, int Cout, int Hout, int Wout,
                  int ks, int stride, int pad, const unsigned short* resid, int relu) {
    dim3 g((unsigned)((Hout * Wout) / 128), (unsigned)(Cout / 32));
    if (ks == 3)
      k_conv_igemm_wmma<3><<<g, 256, 0, stream>>>(X, Cin, Hin, Win, q.w, q.s, q.b,
                                                  Y, Cout, Hout, Wout, stride, pad, resid, relu);
    else
      k_conv_igemm_wmma<1><<<g, 256, 0, stream>>>(X, Cin, Hin, Win, q.w, q.s, q.b,
                                                  Y, Cout, Hout, Wout, stride, pad, resid, relu);
  };
  auto runblk = [&](const QB& q, unsigned short* cur, unsigned short* tmp, int C, int H, int W) {
    conv(cur, C, H, W, q.c1, tmp, C, H, W, 3, 1, 1, nullptr, 1);
    conv(tmp, C, H, W, q.c2, cur, C, H, W, 3, 1, 1, cur, 1);  // relu(x + bn(conv2(y)))
  };
  auto pool = [&](const unsigned short* feat, int H, int W, const float* cA, float srA,
                  const float* cB, float srB, int Ho, int Wo, int C, unsigned short* out) {
    long nv = (long)C * Ho * Wo;
    k_pool_init<<<gb(nv), 256, 0, stream>>>(enc, nv);
    dim3 gs(gb(N), (unsigned)C);
    k_gather_scatter_max<<<gs, 256, 0, stream>>>(feat, H, W, cA, srA, cB, srB, Ho, Wo, enc, N);
    k_pool_decode<<<gb(nv), 256, 0, stream>>>(enc, out, nv);
  };

  // ---- stage 0: convert input f32 -> bf16 (halve HBM bytes up front) ----
  long nIn = 192ll * 512 * 512;
  k_f32_to_bf16<<<gb(nIn), 256, 0, stream>>>(feat_in, X0, nIn);

  // ---- descartes_header (param leaves: blocks@49, down@67) ----
  {
    Q3 down = mkcbr(67, 68, 69, 32, 192, 3);
    conv(X0, 192, 512, 512, down, des0cat, 32, 256, 256, 3, 2, 1, nullptr, 1);
    QB b0 = mkblk(49, 32), b1 = mkblk(55, 32), b2 = mkblk(61, 32);
    runblk(b0, des0cat, tmpB, 32, 256, 256);
    runblk(b1, des0cat, tmpB, 32, 256, 256);
    runblk(b2, des0cat, tmpB, 32, 256, 256);
  }
  // des0 -> cylinder grid (32,1024), sr=0.5 both
  pool(des0cat, 256, 256, dcoord, 0.5f, ccoord, 0.5f, 32, 1024, 32, d2c0);

  // ---- cylinder_header (blocks@13, down@25) ----
  {
    Q3 down = mkcbr(25, 26, 27, 32, 32, 3);
    conv(d2c0, 32, 32, 1024, down, cyl0, 32, 32, 1024, 3, 1, 1, nullptr, 1);
    QB b0 = mkblk(13, 32), b1 = mkblk(19, 32);
    runblk(b0, cyl0, cylT, 32, 32, 1024);
    runblk(b1, cyl0, cylT, 32, 32, 1024);
  }
  // cyl0 -> descartes grid (256,256) into second half of des0cat (free concat)
  pool(cyl0, 32, 1024, ccoord, 0.5f, dcoord, 0.5f, 256, 256, 32, des0cat + 32ull * 65536);

  // header_channel_down 1x1: [64,256,256] -> cat224 ch0..31 (== res0 == des0')
  {
    Q3 q = mkcbr(130, 131, 132, 32, 64, 1);
    conv(des0cat, 64, 256, 256, q, cat224, 32, 256, 256, 1, 1, 0, nullptr, 1);
  }

  // ---- descartes_res1 (blocks@70, down@94) ----
  {
    Q3 down = mkcbr(94, 95, 96, 64, 32, 3);
    conv(cat224, 32, 256, 256, down, des1cat, 64, 128, 128, 3, 2, 1, nullptr, 1);
    QB b0 = mkblk(70, 64), b1 = mkblk(76, 64), b2 = mkblk(82, 64), b3 = mkblk(88, 64);
    runblk(b0, des1cat, tmpB, 64, 128, 128);
    runblk(b1, des1cat, tmpB, 64, 128, 128);
    runblk(b2, des1cat, tmpB, 64, 128, 128);
    runblk(b3, des1cat, tmpB, 64, 128, 128);
  }
  pool(des1cat, 128, 128, dcoord, 0.25f, ccoord, 0.25f, 16, 512, 64, d2c1);

  // ---- cylinder_res1 (blocks@28, down@46) ----
  {
    Q3 down = mkcbr(46, 47, 48, 64, 64, 3);
    conv(d2c1, 64, 16, 512, down, cyl1, 64, 16, 512, 3, 1, 1, nullptr, 1);
    QB b0 = mkblk(28, 64), b1 = mkblk(34, 64), b2 = mkblk(40, 64);
    runblk(b0, cyl1, cylT, 64, 16, 512);
    runblk(b1, cyl1, cylT, 64, 16, 512);
    runblk(b2, cyl1, cylT, 64, 16, 512);
  }
  pool(cyl1, 16, 512, ccoord, 0.25f, dcoord, 0.25f, 128, 128, 64, des1cat + 64ull * 16384);

  // res1_channel_down 1x1: [128,128,128] -> des1d
  {
    Q3 q = mkcbr(139, 140, 141, 64, 128, 1);
    conv(des1cat, 128, 128, 128, q, des1d, 64, 128, 128, 1, 1, 0, nullptr, 1);
  }

  // ---- descartes_res2 (blocks@97, down@127) ----
  {
    Q3 down = mkcbr(127, 128, 129, 128, 64, 3);
    conv(des1d, 64, 128, 128, down, tmpA, 128, 64, 64, 3, 2, 1, nullptr, 1);
    QB b0 = mkblk(97, 128), b1 = mkblk(103, 128), b2 = mkblk(109, 128),
       b3 = mkblk(115, 128), b4 = mkblk(121, 128);
    runblk(b0, tmpA, tmpB, 128, 64, 64);
    runblk(b1, tmpA, tmpB, 128, 64, 64);
    runblk(b2, tmpA, tmpB, 128, 64, 64);
    runblk(b3, tmpA, tmpB, 128, 64, 64);
    runblk(b4, tmpA, tmpB, 128, 64, 64);
  }
  // des2 = cbr(des2_pre + deep_128), add_fuse @4..6
  k_add_f32res<<<gb(128ll * 4096), 256, 0, stream>>>(tmpA, deep, sum2, 128ll * 4096);
  {
    Q3 q = mkcbr(4, 5, 6, 128, 128, 1);
    conv(sum2, 128, 64, 64, q, des2, 128, 64, 64, 1, 1, 0, nullptr, 1);
  }

  // ---- upsample into the concat buffer (free concat) ----
  {
    dim3 g1(gb(65536), 64);
    k_upsample_ac<<<g1, 256, 0, stream>>>(des1d, 128, 128, cat224 + 32ull * 65536, 256, 256);
    dim3 g2(gb(65536), 128);
    k_upsample_ac<<<g2, 256, 0, stream>>>(des2, 64, 64, cat224 + 96ull * 65536, 256, 256);
  }

  // ---- output convs on 224-ch concat ----
  {
    Q3 q1 = mkcbr(133, 134, 135, 128, 224, 3);
    conv(cat224, 224, 256, 256, q1, tmpA, 128, 256, 256, 3, 1, 1, nullptr, 1);
    Q3 q2 = mkcbr(136, 137, 138, 64, 128, 3);
    conv(tmpA, 128, 256, 256, q2, desout, 64, 256, 256, 3, 1, 1, nullptr, 1);
  }

  // ---- final outputs (concatenated flat in return order) ----
  float* out       = (float*)d_out;
  float* o_despt   = out;                              // [64,N]
  float* o_cylpt   = o_despt + 64ll * N;               // [64,N]
  float* o_aux1    = o_cylpt + 64ll * N;               // [3,65536]
  float* o_aux2    = o_aux1 + 3ll * 65536;
  float* o_aux3    = o_aux2 + 3ll * 65536;
  float* o_des2    = o_aux3 + 3ll * 65536;             // [128,4096]

  {
    dim3 g1(gb(N), 64);
    k_grid_to_point<<<g1, 256, 0, stream>>>(desout, 256, 256, dcoord, 0.5f, o_despt, N);
    k_grid_to_point<<<g1, 256, 0, stream>>>(cyl1, 16, 512, ccoord, 0.25f, o_cylpt, N);
    dim3 ga(gb(65536), 3);
    k_aux<<<ga, 256, 0, stream>>>(cat224, 32, F(8), F(7), o_aux1, 65536);
    k_aux<<<ga, 256, 0, stream>>>(cat224 + 32ull * 65536, 64, F(10), F(9), o_aux2, 65536);
    k_aux<<<ga, 256, 0, stream>>>(cat224 + 96ull * 65536, 128, F(12), F(11), o_aux3, 65536);
  }
  k_bf16_to_f32<<<gb(128ll * 4096), 256, 0, stream>>>(des2, o_des2, 128ll * 4096);
}
